// NonLocalBlock_19679540150545
// MI455X (gfx1250) — matrix operand order repaired and checked by reading, not compiled, not run on hardware
//
#include <hip/hip_runtime.h>
#include <hip/hip_bf16.h>

#define DI __device__ __forceinline__

typedef __attribute__((ext_vector_type(16))) __bf16 v16bf;
typedef __attribute__((ext_vector_type(8)))  float  v8f;
typedef __attribute__((ext_vector_type(4)))  int    vi4;

static constexpr int BATCH = 8;
static constexpr int C_IN  = 256;
static constexpr int CI    = 128;
static constexpr int NPIX  = 4096;   // 64*64
static constexpr int MPOOL = 1024;   // 32*32
static constexpr int WIMG  = 64;
static constexpr float BN_EPS_F = 1e-5f;

// ---- workspace layout (bytes) ----
static constexpr size_t OFF_THETA = 0;                       // bf16 [B][N][Ci]   8 MB
static constexpr size_t OFF_PHIF  = OFF_THETA + 8388608;     // bf16 [B][N][Ci]   8 MB
static constexpr size_t OFF_GF    = OFF_PHIF  + 8388608;     // bf16 [B][N][Ci]   8 MB
static constexpr size_t OFF_PHIP  = OFF_GF    + 8388608;     // bf16 [B][Ci][M]   2 MB
static constexpr size_t OFF_GP    = OFF_PHIP  + 2097152;     // bf16 [B][M][Ci]   2 MB
static constexpr size_t OFF_Y     = OFF_GP    + 2097152;     // bf16 [B][N][Ci]   8 MB
static constexpr size_t OFF_G     = OFF_Y     + 8388608;     // f32  [Ci][Ci]    64 KB
static constexpr size_t OFF_SY    = OFF_G     + 65536;       // f32  [Ci]
static constexpr size_t OFF_SC    = OFF_SY    + 512;         // f32  [C]
static constexpr size_t OFF_SH    = OFF_SC    + 1024;        // f32  [C]

// ---- CDNA5 async global->LDS copy (ASYNCcnt-tracked), with safe fallback ----
#if __has_builtin(__builtin_amdgcn_global_load_async_to_lds_b128)
#define HAVE_ASYNC_LDS 1
#else
#define HAVE_ASYNC_LDS 0
#endif

DI void async_copy_b128(const void* g, void* l) {
#if HAVE_ASYNC_LDS
  __builtin_amdgcn_global_load_async_to_lds_b128(
      (__attribute__((address_space(1))) vi4*)const_cast<void*>(g),
      (__attribute__((address_space(3))) vi4*)l, 0, 0);
#else
  *(uint4*)l = *(const uint4*)g;
#endif
}

DI void async_wait0() {
#if HAVE_ASYNC_LDS
#if __has_builtin(__builtin_amdgcn_s_wait_asynccnt)
  __builtin_amdgcn_s_wait_asynccnt(0);
#else
  asm volatile("s_wait_asynccnt 0" ::: "memory");
#endif
#endif
}

// A-matrix (16x32 bf16, wave32) K index of VGPR-pair j for lane group grp.
DI int a_k_of_j(int j, int grp) {
  return (j < 4 ? 2 * j : 2 * (j - 4) + 16) + 8 * grp;
}

DI v8f vzero() {
  v8f z;
#pragma unroll
  for (int i = 0; i < 8; ++i) z[i] = 0.0f;
  return z;
}

DI v8f wmma_bf16(v16bf a, v16bf b, v8f c) {
  return __builtin_amdgcn_wmma_f32_16x16x32_bf16(false, a, false, b, (short)0, c,
                                                 false, false);
}

// ============================================================================
// Kernel 1: 1x1 conv projection.  out[b][n][ci] = sum_c x[b][c][n]*w[ci][c]+bias
// Block = 256 thr (8 waves) computes a 64(n) x 128(ci) tile. K=256 in 8 steps.
// ============================================================================
__global__ void proj_kernel(const float* __restrict__ x, const float* __restrict__ w,
                            const float* __restrict__ bias, __bf16* __restrict__ out) {
  __shared__ __bf16 ws[32][CI + 8];   // W^T k-slab: ws[c_local][ci]
  const int b     = blockIdx.x >> 6;
  const int nbase = (blockIdx.x & 63) * 64;
  const int tid   = threadIdx.x;
  const int wid = tid >> 5, lane = tid & 31;
  const int m = lane & 15, grp = lane >> 4;
  const int nsub   = wid & 3;            // which 16-row n subtile
  const int cstart = (wid >> 2) * 64;    // 4 ci subtiles per wave
  const int row_n  = nbase + nsub * 16 + m;

  v8f acc[4];
#pragma unroll
  for (int t = 0; t < 4; ++t) acc[t] = vzero();

  for (int ks = 0; ks < 8; ++ks) {
    const int c0 = ks * 32;
    __syncthreads();
    for (int i = tid; i < 32 * CI; i += 256) {     // stage transposed weights
      int cc = i >> 7, ci = i & 127;
      ws[cc][ci] = (__bf16)w[ci * C_IN + c0 + cc];
    }
    __syncthreads();
    v16bf a;
#pragma unroll
    for (int j = 0; j < 8; ++j) {                  // A = X^T tile, fp32->bf16
      int kk = a_k_of_j(j, grp);
      size_t base = ((size_t)(b * C_IN + c0 + kk)) * NPIX + row_n;
      a[2 * j]     = (__bf16)x[base];
      a[2 * j + 1] = (__bf16)x[base + NPIX];
    }
#pragma unroll
    for (int t = 0; t < 4; ++t) {
      int ci0 = cstart + 16 * t;
      v16bf bb;                                    // B col = lane&15, elems over K (table order)
#pragma unroll
      for (int e = 0; e < 16; ++e) bb[e] = ws[(e < 8 ? e : e + 8) + 8 * grp][ci0 + m];
      acc[t] = wmma_bf16(a, bb, acc[t]);
    }
  }
#pragma unroll
  for (int t = 0; t < 4; ++t) {
    int cicol = cstart + 16 * t + m;               // D col = lane&15
    float bv = bias[cicol];
#pragma unroll
    for (int v = 0; v < 8; ++v) {
      int row = nsub * 16 + v + 8 * grp;           // D row = v + 8*grp
      out[((size_t)b * NPIX + nbase + row) * CI + cicol] = (__bf16)(acc[t][v] + bv);
    }
  }
}

// ============================================================================
// Kernel 2: 2x2 maxpool.  g -> [b][m][ci], phi -> [b][ci][m] (transposed)
// ============================================================================
__global__ void pool_kernel(const __bf16* __restrict__ phi_full,
                            const __bf16* __restrict__ g_full,
                            __bf16* __restrict__ phi_p, __bf16* __restrict__ g_p) {
  const int T = BATCH * MPOOL * CI;
  int idx = blockIdx.x * 256 + threadIdx.x;
  if (idx < T) {                                   // g: out [b][m][ci]
    int ci = idx & 127, mm = (idx >> 7) & 1023, b = idx >> 17;
    int mh = mm >> 5, mw = mm & 31;
    int n0 = (2 * mh) * WIMG + 2 * mw;
    size_t base = ((size_t)b * NPIX) * CI + ci;
    float v0 = (float)g_full[base + (size_t)n0 * CI];
    float v1 = (float)g_full[base + (size_t)(n0 + 1) * CI];
    float v2 = (float)g_full[base + (size_t)(n0 + WIMG) * CI];
    float v3 = (float)g_full[base + (size_t)(n0 + WIMG + 1) * CI];
    g_p[idx] = (__bf16)fmaxf(fmaxf(v0, v1), fmaxf(v2, v3));
  } else if (idx < 2 * T) {                        // phi: out [b][ci][m]
    int j = idx - T;
    int mm = j & 1023, ci = (j >> 10) & 127, b = j >> 17;
    int mh = mm >> 5, mw = mm & 31;
    int n0 = (2 * mh) * WIMG + 2 * mw;
    size_t base = ((size_t)b * NPIX) * CI + ci;
    float v0 = (float)phi_full[base + (size_t)n0 * CI];
    float v1 = (float)phi_full[base + (size_t)(n0 + 1) * CI];
    float v2 = (float)phi_full[base + (size_t)(n0 + WIMG) * CI];
    float v3 = (float)phi_full[base + (size_t)(n0 + WIMG + 1) * CI];
    phi_p[j] = (__bf16)fmaxf(fmaxf(v0, v1), fmaxf(v2, v3));
  }
}

// ============================================================================
// Kernel 3: attention.  Per block: 16 query rows, all M=1024 keys.
//   S = theta @ phi  -> LDS (bf16) -> row softmax (unnormalized) -> y = P @ g
// ============================================================================
__global__ void attn_kernel(const __bf16* __restrict__ theta,
                            const __bf16* __restrict__ phi_p,
                            const __bf16* __restrict__ g_p,
                            __bf16* __restrict__ y) {
  __shared__ __bf16 sp[16][MPOOL + 8];             // logits / probs
  __shared__ float inv_sum[16];
  const int b     = blockIdx.x >> 8;
  const int nbase = (blockIdx.x & 255) * 16;
  const int tid = threadIdx.x;
  const int wid = tid >> 5, lane = tid & 31;
  const int m = lane & 15, grp = lane >> 4;

  // ---- Phase 1: logits.  A frags (theta rows) reused across all 64 m-tiles.
  v16bf afr[4];
  {
    size_t tbase = ((size_t)b * NPIX + nbase + m) * CI;
#pragma unroll
    for (int ks = 0; ks < 4; ++ks)
#pragma unroll
      for (int j = 0; j < 8; ++j) {
        int kk = ks * 32 + a_k_of_j(j, grp);
        afr[ks][2 * j]     = theta[tbase + kk];
        afr[ks][2 * j + 1] = theta[tbase + kk + 1];
      }
  }
  for (int t = 0; t < 8; ++t) {                    // 8 of 64 m-tiles per wave
    int m0 = (wid + 8 * t) * 16;
    v8f acc = vzero();
#pragma unroll
    for (int ks = 0; ks < 4; ++ks) {
      const __bf16* src = &phi_p[((size_t)b * CI + ks * 32) * MPOOL + m0 + m];
      v16bf bb;
#pragma unroll
      for (int e = 0; e < 16; ++e) bb[e] = src[(size_t)((e < 8 ? e : e + 8) + 8 * grp) * MPOOL];
      acc = wmma_bf16(afr[ks], bb, acc);
    }
#pragma unroll
    for (int v = 0; v < 8; ++v) sp[v + 8 * grp][m0 + m] = (__bf16)acc[v];
  }
  __syncthreads();

  // ---- Phase 2: softmax, 16 threads per row, exp kept unnormalized.
  {
    int row = tid >> 4, slot = tid & 15;
    float mx = -3.4e38f;
    for (int j = 0; j < 64; ++j)
      mx = fmaxf(mx, (float)sp[row][slot * 64 + j]);
#pragma unroll
    for (int d = 1; d < 16; d <<= 1) mx = fmaxf(mx, __shfl_xor(mx, d, 32));
    float sum = 0.0f;
    for (int j = 0; j < 64; ++j) {
      float e = __expf((float)sp[row][slot * 64 + j] - mx);
      sp[row][slot * 64 + j] = (__bf16)e;
      sum += e;
    }
#pragma unroll
    for (int d = 1; d < 16; d <<= 1) sum += __shfl_xor(sum, d, 32);
    if (slot == 0) inv_sum[row] = 1.0f / sum;
  }
  __syncthreads();

  // ---- Phase 3: y = P @ g.  Each wave owns one 16-wide ci stripe, K=1024.
  {
    int ci0 = wid * 16;
    v8f acc = vzero();
    for (int ks = 0; ks < 32; ++ks) {
      int k0 = ks * 32;
      v16bf a;
#pragma unroll
      for (int j = 0; j < 8; ++j) {
        int kk = k0 + a_k_of_j(j, grp);
        a[2 * j]     = sp[m][kk];
        a[2 * j + 1] = sp[m][kk + 1];
      }
      const __bf16* src = &g_p[((size_t)b * MPOOL + k0) * CI + ci0 + m];
      v16bf bb;
#pragma unroll
      for (int e = 0; e < 16; ++e) bb[e] = src[(size_t)((e < 8 ? e : e + 8) + 8 * grp) * CI];
      acc = wmma_bf16(a, bb, acc);
    }
#pragma unroll
    for (int v = 0; v < 8; ++v) {
      int row = v + 8 * grp;
      y[((size_t)b * NPIX + nbase + row) * CI + ci0 + m] =
          (__bf16)(acc[v] * inv_sum[row]);
    }
  }
}

// ============================================================================
// Kernel 4: BN stats base.  Sy[ci] = sum_p y, G = Y^T Y via WMMA + atomics.
// Block handles 256 rows of Y[32768][128]; tiles staged with async LDS copies.
// ============================================================================
__global__ void stats_kernel(const __bf16* __restrict__ y, float* __restrict__ G,
                             float* __restrict__ Sy) {
  __shared__ __bf16 ys[32][CI + 8];
  const int tid = threadIdx.x;
  const int r0 = blockIdx.x * 256;
  {
    int ci = tid & 127, half = tid >> 7;
    float s = 0.0f;
    for (int r = r0 + half; r < r0 + 256; r += 2) s += (float)y[(size_t)r * CI + ci];
    atomicAdd(&Sy[ci], s);
  }
  const int wid = tid >> 5, lane = tid & 31;
  const int m = lane & 15, grp = lane >> 4;
  v8f acc[8];
#pragma unroll
  for (int jt = 0; jt < 8; ++jt) acc[jt] = vzero();

  for (int step = 0; step < 8; ++step) {
    __syncthreads();
    // stage 32x128 bf16 = 512 x 16B chunks; 2 async B128 copies per thread
#pragma unroll
    for (int i = tid; i < 512; i += 256) {
      int rr = i >> 4, c8 = (i & 15) * 8;
      async_copy_b128(&y[(size_t)(r0 + step * 32 + rr) * CI + c8], &ys[rr][c8]);
    }
    async_wait0();
    __syncthreads();
    v16bf a;                                       // A = Y^T (transposed read)
#pragma unroll
    for (int j = 0; j < 8; ++j) {
      int kk = a_k_of_j(j, grp);
      a[2 * j]     = ys[kk][wid * 16 + m];
      a[2 * j + 1] = ys[kk + 1][wid * 16 + m];
    }
#pragma unroll
    for (int jt = 0; jt < 8; ++jt) {
      v16bf bb;
#pragma unroll
      for (int e = 0; e < 16; ++e) bb[e] = ys[(e < 8 ? e : e + 8) + 8 * grp][jt * 16 + m];
      acc[jt] = wmma_bf16(a, bb, acc[jt]);
    }
  }
#pragma unroll
  for (int jt = 0; jt < 8; ++jt)
#pragma unroll
    for (int v = 0; v < 8; ++v)
      atomicAdd(&G[(size_t)(wid * 16 + v + 8 * grp) * CI + jt * 16 + m], acc[jt][v]);
}

__global__ void zero_kernel(float* __restrict__ G, float* __restrict__ Sy) {
  int idx = blockIdx.x * 256 + threadIdx.x;
  if (idx < CI * CI) G[idx] = 0.0f;
  else if (idx < CI * CI + CI) Sy[idx - CI * CI] = 0.0f;
}

// ============================================================================
// Kernel 5: per-channel BN scale/shift from (Sy, G).
//   mean_o = w_o.mu + b_o ;  var_o = (w_o^T G w_o)/P - (w_o.mu)^2
// ============================================================================
__global__ void bnstat_kernel(const float* __restrict__ G, const float* __restrict__ Sy,
                              const float* __restrict__ Ww, const float* __restrict__ Wb,
                              const float* __restrict__ gamma, const float* __restrict__ beta,
                              float* __restrict__ scale, float* __restrict__ shift) {
  const int o = threadIdx.x;                        // 256 threads, 1 block
  const float invP = 1.0f / (float)(BATCH * NPIX);
  float wmu = 0.0f;
  for (int i = 0; i < CI; ++i) wmu += Ww[o * CI + i] * Sy[i];
  wmu *= invP;
  float q = 0.0f;
  for (int i = 0; i < CI; ++i) {
    float wi = Ww[o * CI + i];
    float t = 0.0f;
    for (int j = 0; j < CI; ++j) t += G[i * CI + j] * Ww[o * CI + j];
    q += wi * t;
  }
  q *= invP;
  float var = q - wmu * wmu;
  float mean = wmu + Wb[o];
  float sc = gamma[o] * rsqrtf(var + BN_EPS_F);
  scale[o] = sc;
  shift[o] = beta[o] - mean * sc;
}

// ============================================================================
// Kernel 6: out = scale*(W@y) + shift + x, D rows = out-channel,
// D cols = pixels (coalesced fp32 stores). Block: 64 pixels x 256 channels.
// ============================================================================
__global__ void out_kernel(const __bf16* __restrict__ y, const float* __restrict__ Ww,
                           const float* __restrict__ x, const float* __restrict__ scale,
                           const float* __restrict__ shift, float* __restrict__ out) {
  __shared__ __bf16 ys[64][CI + 8];
  const int b     = blockIdx.x >> 6;
  const int nbase = (blockIdx.x & 63) * 64;
  const int tid = threadIdx.x;
  // stage 64x128 bf16 = 1024 x 16B chunks; 4 async B128 copies per thread
#pragma unroll
  for (int i = tid; i < 1024; i += 256) {
    int nn = i >> 4, c8 = (i & 15) * 8;
    async_copy_b128(&y[((size_t)b * NPIX + nbase + nn) * CI + c8], &ys[nn][c8]);
  }
  async_wait0();
  __syncthreads();
  const int wid = tid >> 5, lane = tid & 31;
  const int m = lane & 15, grp = lane >> 4;

  v16bf af[2][4];                                  // A = W rows (bf16), 2 o-stripes
#pragma unroll
  for (int s = 0; s < 2; ++s) {
    int o0 = (wid + 8 * s) * 16;
#pragma unroll
    for (int ks = 0; ks < 4; ++ks)
#pragma unroll
      for (int j = 0; j < 8; ++j) {
        int kk = ks * 32 + a_k_of_j(j, grp);
        af[s][ks][2 * j]     = (__bf16)Ww[(o0 + m) * CI + kk];
        af[s][ks][2 * j + 1] = (__bf16)Ww[(o0 + m) * CI + kk + 1];
      }
  }
  v8f acc[2][4];
#pragma unroll
  for (int s = 0; s < 2; ++s)
#pragma unroll
    for (int ns = 0; ns < 4; ++ns) acc[s][ns] = vzero();

#pragma unroll
  for (int ns = 0; ns < 4; ++ns) {
#pragma unroll
    for (int ks = 0; ks < 4; ++ks) {
      v16bf bb;                                    // B = y^T : col = lane&15, elems over K ci
#pragma unroll
      for (int e = 0; e < 16; ++e) bb[e] = ys[ns * 16 + m][ks * 32 + (e < 8 ? e : e + 8) + 8 * grp];
      acc[0][ns] = wmma_bf16(af[0][ks], bb, acc[0][ns]);
      acc[1][ns] = wmma_bf16(af[1][ks], bb, acc[1][ns]);
    }
  }
#pragma unroll
  for (int s = 0; s < 2; ++s) {
    int o0 = (wid + 8 * s) * 16;
#pragma unroll
    for (int v = 0; v < 8; ++v) {
      int orow = o0 + v + 8 * grp;
      float sc = scale[orow], sh = shift[orow];
#pragma unroll
      for (int ns = 0; ns < 4; ++ns) {
        size_t oi = ((size_t)b * C_IN + orow) * NPIX + nbase + ns * 16 + m;
        out[oi] = acc[s][ns][v] * sc + sh + x[oi];
      }
    }
  }
}

// ============================================================================
extern "C" void kernel_launch(void* const* d_in, const int* in_sizes, int n_in,
                              void* d_out, int out_size, void* d_ws, size_t ws_size,
                              hipStream_t stream) {
  const float* x       = (const float*)d_in[0];
  const float* theta_w = (const float*)d_in[1];
  const float* theta_b = (const float*)d_in[2];
  const float* phi_w   = (const float*)d_in[3];
  const float* phi_b   = (const float*)d_in[4];
  const float* g_w     = (const float*)d_in[5];
  const float* g_b     = (const float*)d_in[6];
  const float* W_w     = (const float*)d_in[7];
  const float* W_b     = (const float*)d_in[8];
  const float* bn_g    = (const float*)d_in[9];
  const float* bn_b    = (const float*)d_in[10];
  float* out = (float*)d_out;

  char* ws = (char*)d_ws;
  __bf16* theta_buf = (__bf16*)(ws + OFF_THETA);
  __bf16* phif      = (__bf16*)(ws + OFF_PHIF);
  __bf16* gf        = (__bf16*)(ws + OFF_GF);
  __bf16* phip      = (__bf16*)(ws + OFF_PHIP);
  __bf16* gp        = (__bf16*)(ws + OFF_GP);
  __bf16* ybuf      = (__bf16*)(ws + OFF_Y);
  float*  Gbuf      = (float*)(ws + OFF_G);
  float*  Sybuf     = (float*)(ws + OFF_SY);
  float*  scbuf     = (float*)(ws + OFF_SC);
  float*  shbuf     = (float*)(ws + OFF_SH);

  proj_kernel<<<512, 256, 0, stream>>>(x, theta_w, theta_b, theta_buf);
  proj_kernel<<<512, 256, 0, stream>>>(x, phi_w, phi_b, phif);
  proj_kernel<<<512, 256, 0, stream>>>(x, g_w, g_b, gf);
  pool_kernel<<<8192, 256, 0, stream>>>(phif, gf, phip, gp);
  attn_kernel<<<2048, 256, 0, stream>>>(theta_buf, phip, gp, ybuf);
  zero_kernel<<<65, 256, 0, stream>>>(Gbuf, Sybuf);
  stats_kernel<<<128, 256, 0, stream>>>(ybuf, Gbuf, Sybuf);
  bnstat_kernel<<<1, 256, 0, stream>>>(Gbuf, Sybuf, W_w, W_b, bn_g, bn_b, scbuf, shbuf);
  out_kernel<<<512, 256, 0, stream>>>(ybuf, W_w, x, scbuf, shbuf, out);
}
